// VarianceAdaptor_31774168056188
// MI455X (gfx1250) — matrix operand order repaired and checked by reading, not compiled, not run on hardware
//
#include <hip/hip_runtime.h>

// ---------------------------------------------------------------------------
// Problem constants (match reference; all fixed)
// ---------------------------------------------------------------------------
#define BB   32
#define LL   512            // sequence length (2^9)
#define HH   512            // hidden / channels (2^9)
#define FF   512            // filter channels == GEMM N
#define TT   4096
#define NBM1 255            // NBINS-1 bins
#define MROWS (BB*LL)       // 16384
#define EPSV 1e-8f

// GEMM tiling: block 256 thr = 8 waves; block tile 128x128; wave tile 64x32
#define BM 128
#define BN 128
#define KT 32
#define KTP 40              // padded LDS row stride (ushorts); 80B rows, 16B-aligned frags

typedef __bf16 bf16_t;
typedef __attribute__((ext_vector_type(16))) bf16_t bf16x16;
typedef __attribute__((ext_vector_type(8)))  float  floatx8;

__device__ __forceinline__ unsigned short f2bf(float f) {
    unsigned int u = __float_as_uint(f);
    unsigned int r = u + 0x7FFFu + ((u >> 16) & 1u);
    return (unsigned short)(r >> 16);
}
__device__ __forceinline__ float bf2f(unsigned short h) {
    return __uint_as_float(((unsigned int)h) << 16);
}

// ---------------------------------------------------------------------------
// f32 -> bf16 convert (grid-stride)
// ---------------------------------------------------------------------------
__global__ void cvt_bf16_kernel(const float* __restrict__ in,
                                unsigned short* __restrict__ out, long n) {
    long i = (long)blockIdx.x * blockDim.x + threadIdx.x;
    long stride = (long)gridDim.x * blockDim.x;
    for (; i < n; i += stride) out[i] = f2bf(in[i]);
}

// f32 (KTOT x FF row-major) -> bf16 transposed (FF x KTOT row-major).
// Tiled 32x32 through LDS; block 256 threads.
template<int KTOT>
__global__ __launch_bounds__(256) void cvtT_bf16_kernel(
    const float* __restrict__ in, unsigned short* __restrict__ out)
{
    __shared__ unsigned short tile[32][33];
    const int k0 = blockIdx.y * 32;
    const int n0 = blockIdx.x * 32;
    const int tx = threadIdx.x & 31;
    const int ty = threadIdx.x >> 5;     // 0..7
#pragma unroll
    for (int r = 0; r < 32; r += 8)
        tile[ty + r][tx] = f2bf(in[(long)(k0 + ty + r) * FF + n0 + tx]);
    __syncthreads();
#pragma unroll
    for (int r = 0; r < 32; r += 8)
        out[(long)(n0 + ty + r) * KTOT + k0 + tx] = tile[tx][ty + r];
}

// out_bf16 = bf16(x + p)  (grid-stride)
__global__ void addx_bf16_kernel(const float* __restrict__ x,
                                 const float* __restrict__ p,
                                 unsigned short* __restrict__ out, long n) {
    long i = (long)blockIdx.x * blockDim.x + threadIdx.x;
    long stride = (long)gridDim.x * blockDim.x;
    for (; i < n; i += stride) out[i] = f2bf(x[i] + p[i]);
}

// ---------------------------------------------------------------------------
// WMMA bf16 GEMM:  C[MROWS,FF] = relu(A @ W^T + bias)
//   A  : MROWS x KTOT row-major bf16 (CONV==1: im2col view of (BB,LL,HH),
//        k=3 pad=1; segment size == HH so address is linear in ko and only
//        the row-validity predicate depends on the k-segment).
//   BwT: FF x KTOT row-major bf16 (pre-transposed weights) -> B staging is
//        identical to A staging: 2x global b128 + 2x ds b128 per thread.
//   OBF: write bf16 (next-layer A operand) else f32.
// Wave tile 64x32 (4x2 WMMA tiles); waves arranged 2(M) x 4(N).
// Fragments load as 2x ds_load_b128 each (16B-aligned thanks to KTP=40).
// ---------------------------------------------------------------------------
template<int CONV, int KTOT, int OBF>
__global__ __launch_bounds__(256) void gemm_bf16_kernel(
    const unsigned short* __restrict__ A,
    const unsigned short* __restrict__ BwT,
    const float* __restrict__ bias,
    float* __restrict__ outF,
    unsigned short* __restrict__ outBF)
{
    __shared__ unsigned short As[BM * KTP];
    __shared__ unsigned short BsT[BN * KTP];

    const int tid  = threadIdx.x;
    const int lane = tid & 31;
    const int wid  = tid >> 5;
    const int wm   = wid >> 2;         // 2 waves along M (tile 64)
    const int wn   = wid & 3;          // 4 waves along N (tile 32)
    const int blockM = blockIdx.y * BM;
    const int blockN = blockIdx.x * BN;

    floatx8 acc[4][2] = {};

    // ---- staging geometry: one 16-elem half-row per thread (A and B) ----
    const int srow  = tid >> 1;        // 0..127
    const int shalf = (tid & 1) << 4;  // 0 or 16
    const int gr    = blockM + srow;
    const unsigned short* aBase;
    int ll_ = 0;
    if (CONV) {
        int b_ = gr >> 9;              // gr / LL
        ll_    = gr & (LL - 1);
        aBase  = A + ((long)(b_ * LL + ll_ - 1) * HH + shalf);
    } else {
        aBase  = A + ((long)gr * KTOT + shalf);
    }
    const unsigned short* bBase = BwT + ((long)(blockN + srow) * KTOT + shalf);

    const int mlane = lane & 15;
    const int khalf = lane >> 4;       // 0/1
    const int kb    = khalf * 8;

    for (int ko = 0; ko < KTOT; ko += KT) {
        // ---- stage A tile (128 x 32) ----
        {
            bool valid = true;
            if (CONV) valid = ((unsigned)(ll_ + (ko >> 9) - 1) < (unsigned)LL);
            const unsigned short* src = aBase + ko;
            uint4 lo = {0, 0, 0, 0}, hi = {0, 0, 0, 0};
            if (valid) {
                lo = *(const uint4*)src;
                hi = *(const uint4*)(src + 8);
            }
            if (ko + KT < KTOT) __builtin_prefetch(aBase + ko + KT, 0, 3);
            uint4* d = (uint4*)&As[srow * KTP + shalf];
            d[0] = lo; d[1] = hi;
        }
        // ---- stage B tile (128 x 32) ----
        {
            const unsigned short* src = bBase + ko;
            uint4 lo = *(const uint4*)src;
            uint4 hi = *(const uint4*)(src + 8);
            if (ko + KT < KTOT) __builtin_prefetch(src + KT, 0, 3);
            uint4* d = (uint4*)&BsT[srow * KTP + shalf];
            d[0] = lo; d[1] = hi;
        }
        __syncthreads();

        // ---- fragments + WMMA (4 M-tiles x 2 N-tiles) ----
        union Frag { bf16x16 v; uint4 q[2]; };
        Frag fa[4], fb[2];
#pragma unroll
        for (int mi = 0; mi < 4; ++mi) {
            const unsigned short* base = &As[(wm * 64 + mi * 16 + mlane) * KTP + kb];
            fa[mi].q[0] = *(const uint4*)base;          // K pairs kb+0..kb+6
            fa[mi].q[1] = *(const uint4*)(base + 16);   // K pairs 16+kb..16+kb+6
        }
#pragma unroll
        for (int ni = 0; ni < 2; ++ni) {
            const unsigned short* base = &BsT[(wn * 32 + ni * 16 + mlane) * KTP + kb];
            fb[ni].q[0] = *(const uint4*)base;          // K pairs kb+0..kb+6
            fb[ni].q[1] = *(const uint4*)(base + 16);   // K pairs 16+kb..16+kb+6
        }
#pragma unroll
        for (int mi = 0; mi < 4; ++mi)
#pragma unroll
            for (int ni = 0; ni < 2; ++ni)
                acc[mi][ni] = __builtin_amdgcn_wmma_f32_16x16x32_bf16(
                    false, fa[mi].v, false, fb[ni].v,
                    (short)0, acc[mi][ni], false, false);
        __syncthreads();
    }

    // ---- epilogue: bias + relu ----
    const int rowHi = khalf * 8;
#pragma unroll
    for (int mi = 0; mi < 4; ++mi) {
#pragma unroll
        for (int ni = 0; ni < 2; ++ni) {
            int gcol = blockN + wn * 32 + ni * 16 + mlane;
            float bv = bias[gcol];
#pragma unroll
            for (int r = 0; r < 8; ++r) {
                int grow = blockM + wm * 64 + mi * 16 + rowHi + r;
                float v = acc[mi][ni][r] + bv;
                v = v > 0.f ? v : 0.f;
                long o = (long)grow * FF + gcol;
                if (OBF) outBF[o] = f2bf(v);
                else     outF[o]  = v;
            }
        }
    }
}

// ---------------------------------------------------------------------------
// RMS norm over channel dim (input already relu'd, f32) -> bf16
// out = g * h / (sqrt(mean(h^2)) + eps).  One block per row, C = 512.
// ---------------------------------------------------------------------------
__global__ __launch_bounds__(256) void rms_kernel(
    const float* __restrict__ h, const float* __restrict__ g,
    unsigned short* __restrict__ out)
{
    __shared__ float red[8];
    const int row = blockIdx.x;
    const float* hr = h + (long)row * FF;
    float ss = 0.f;
    for (int c = threadIdx.x; c < FF; c += 256) { float v = hr[c]; ss += v * v; }
    for (int off = 16; off; off >>= 1) ss += __shfl_down(ss, off);
    int lane = threadIdx.x & 31, wid = threadIdx.x >> 5;
    if (lane == 0) red[wid] = ss;
    __syncthreads();
    if (wid == 0) {
        float s = (lane < 8) ? red[lane] : 0.f;
        for (int off = 4; off; off >>= 1) s += __shfl_down(s, off);
        if (lane == 0) red[0] = s;
    }
    __syncthreads();
    float inv = 1.0f / (sqrtf(red[0] / (float)FF) + EPSV);
    unsigned short* orow = out + (long)row * FF;
    for (int c = threadIdx.x; c < FF; c += 256) orow[c] = f2bf(g[c] * hr[c] * inv);
}

// ---------------------------------------------------------------------------
// Final linear F->1 + bias, masked.  One wave per row (8 rows/block).
// ---------------------------------------------------------------------------
__global__ __launch_bounds__(256) void dot_mask_kernel(
    const unsigned short* __restrict__ h, const float* __restrict__ wl,
    const float* __restrict__ bl, const unsigned char* __restrict__ mask,
    float* __restrict__ out)
{
    int wid = threadIdx.x >> 5, lane = threadIdx.x & 31;
    int row = blockIdx.x * 8 + wid;
    const unsigned short* hr = h + (long)row * FF;
    float s = 0.f;
#pragma unroll
    for (int j = 0; j < 16; ++j) {
        int c = lane * 16 + j;
        s += bf2f(hr[c]) * wl[c];
    }
    for (int off = 16; off; off >>= 1) s += __shfl_down(s, off);
    if (lane == 0) out[row] = mask[row] ? 0.0f : (s + bl[0]);
}

// ---------------------------------------------------------------------------
// xo = x + pitch_emb[bucket(pt)] + energy_emb[bucket(et)] + sp_proj
// One block per (b,l) row.  searchsorted(bins, v, 'left') via binary search.
// ---------------------------------------------------------------------------
__global__ __launch_bounds__(256) void compose_xo_kernel(
    const float* __restrict__ x, const float* __restrict__ sp,
    const float* __restrict__ pt, const float* __restrict__ et,
    const float* __restrict__ pbins, const float* __restrict__ ebins,
    const float* __restrict__ pemb, const float* __restrict__ eemb,
    float* __restrict__ xo)
{
    __shared__ int sidx[2];
    const int row = blockIdx.x;
    if (threadIdx.x < 2) {
        const float* bins = threadIdx.x ? ebins : pbins;
        float v = threadIdx.x ? et[row] : pt[row];
        int lo = 0, hi = NBM1;
        while (lo < hi) { int mid = (lo + hi) >> 1; if (bins[mid] < v) lo = mid + 1; else hi = mid; }
        sidx[threadIdx.x] = lo;
    }
    __syncthreads();
    const float* per = pemb + (long)sidx[0] * HH;
    const float* eer = eemb + (long)sidx[1] * HH;
    const float* xr  = x  + (long)row * HH;
    const float* spr = sp + (long)row * HH;
    float* orow = xo + (long)row * HH;
    for (int c = threadIdx.x; c < HH; c += 256)
        orow[c] = xr[c] + per[c] + eer[c] + spr[c];
}

// ---------------------------------------------------------------------------
// Inclusive cumsum of durations per batch (L=512). One block (512 thr)/batch.
// Also writes mel_len (float) to output tail.
// ---------------------------------------------------------------------------
__global__ __launch_bounds__(512) void cumsum_kernel(
    const int* __restrict__ dur, int* __restrict__ cum,
    float* __restrict__ melOut)
{
    __shared__ int s[LL];
    const int b = blockIdx.x, t = threadIdx.x;
    s[t] = dur[b * LL + t];
    __syncthreads();
    for (int off = 1; off < LL; off <<= 1) {
        int v = (t >= off) ? s[t - off] : 0;
        __syncthreads();
        s[t] += v;
        __syncthreads();
    }
    cum[b * LL + t] = s[t];
    if (t == LL - 1) melOut[b] = (float)s[t];
}

// ---------------------------------------------------------------------------
// Length regulate: out[b,t,:] = (t < mel_len) ? xo[b, idx(t), :] : 0
// idx = min(searchsorted_right(cum, t), L-1).  One block per (b,t).
// ---------------------------------------------------------------------------
__global__ __launch_bounds__(256) void lenreg_kernel(
    const float* __restrict__ xo, const int* __restrict__ cum,
    float* __restrict__ out)
{
    const int b = blockIdx.x >> 12;          // / TT
    const int t = blockIdx.x & (TT - 1);
    const int* cr = cum + b * LL;
    const int total = cr[LL - 1];
    int lo = 0, hi = LL;
    while (lo < hi) { int mid = (lo + hi) >> 1; if (cr[mid] <= t) lo = mid + 1; else hi = mid; }
    int idx = lo < LL ? lo : LL - 1;
    const bool valid = t < total;
    const float* src = xo + ((long)(b * LL + idx)) * HH;
    float* dst = out + ((long)b * TT + t) * (long)HH;
    for (int c = threadIdx.x; c < HH; c += 256)
        dst[c] = valid ? src[c] : 0.0f;
}

// ---------------------------------------------------------------------------
// Host launcher
// ---------------------------------------------------------------------------
enum {
    IN_X = 0, IN_EMB, IN_MASK, IN_DURT, IN_PITT, IN_ENGT, IN_MAXLEN,
    IN_DUR_W1, IN_DUR_B1, IN_DUR_G1, IN_DUR_W2, IN_DUR_B2, IN_DUR_G2, IN_DUR_WL, IN_DUR_BL,
    IN_PIT_W1, IN_PIT_B1, IN_PIT_G1, IN_PIT_W2, IN_PIT_B2, IN_PIT_G2, IN_PIT_WL, IN_PIT_BL,
    IN_ENG_W1, IN_ENG_B1, IN_ENG_G1, IN_ENG_W2, IN_ENG_B2, IN_ENG_G2, IN_ENG_WL, IN_ENG_BL,
    IN_PP_W1, IN_PP_B1, IN_PP_W2, IN_PP_B2,
    IN_EP_W1, IN_EP_B1, IN_EP_W2, IN_EP_B2,
    IN_SP_W1, IN_SP_B1, IN_SP_W2, IN_SP_B2,
    IN_PBINS, IN_EBINS, IN_PEMB, IN_EEMB
};

extern "C" void kernel_launch(void* const* d_in, const int* in_sizes, int n_in,
                              void* d_out, int out_size, void* d_ws, size_t ws_size,
                              hipStream_t stream) {
    (void)in_sizes; (void)n_in; (void)out_size; (void)ws_size;

    const long NACT = (long)MROWS * HH;            // 8,388,608
    const size_t SZ_BF  = (size_t)NACT * 2;        // 16 MiB
    const size_t SZ_F32 = (size_t)NACT * 4;        // 32 MiB
    const size_t SZ_PW  = (size_t)HH * FF * 2;     // 512 KiB
    const size_t SZ_CW  = (size_t)3 * HH * FF * 2; // 1.5 MiB

    char* ws = (char*)d_ws;
    unsigned short* xbf  = (unsigned short*)(ws + 0 * SZ_BF);
    unsigned short* ebf  = (unsigned short*)(ws + 1 * SZ_BF);
    unsigned short* bufA = (unsigned short*)(ws + 2 * SZ_BF);
    unsigned short* bufB = (unsigned short*)(ws + 3 * SZ_BF);
    float* accF = (float*)(ws + 4 * SZ_BF);
    float* xoF  = (float*)(ws + 4 * SZ_BF + SZ_F32);
    char*  wb   = ws + 4 * SZ_BF + 2 * SZ_F32;
    unsigned short* ppw1 = (unsigned short*)(wb + 0 * SZ_PW);
    unsigned short* ppw2 = (unsigned short*)(wb + 1 * SZ_PW);
    unsigned short* epw1 = (unsigned short*)(wb + 2 * SZ_PW);
    unsigned short* epw2 = (unsigned short*)(wb + 3 * SZ_PW);
    unsigned short* spw1 = (unsigned short*)(wb + 4 * SZ_PW);
    unsigned short* spw2 = (unsigned short*)(wb + 5 * SZ_PW);
    char* cb = wb + 6 * SZ_PW;
    unsigned short* durw1 = (unsigned short*)(cb + 0 * SZ_CW);
    unsigned short* durw2 = (unsigned short*)(cb + 1 * SZ_CW);
    unsigned short* pitw1 = (unsigned short*)(cb + 2 * SZ_CW);
    unsigned short* pitw2 = (unsigned short*)(cb + 3 * SZ_CW);
    unsigned short* engw1 = (unsigned short*)(cb + 4 * SZ_CW);
    unsigned short* engw2 = (unsigned short*)(cb + 5 * SZ_CW);
    int* cum = (int*)(cb + 6 * SZ_CW);

    float* dout    = (float*)d_out;
    float* outMel0 = dout;                                   // (B,T,H)
    float* outPit  = dout + (long)BB * TT * HH;              // (B,L)
    float* outEng  = outPit + MROWS;
    float* outDur  = outEng + MROWS;
    float* outMelLen = outDur + MROWS;                       // (B,)

    const float* fx = (const float*)d_in[IN_X];
    const unsigned char* mask = (const unsigned char*)d_in[IN_MASK];
    const int* durT = (const int*)d_in[IN_DURT];

    // weights: convert + transpose to N-major (FF x KTOT)
    dim3 tgP(FF / 32, HH / 32);        // 512x512 proj weights
    dim3 tgC(FF / 32, (3 * HH) / 32);  // 1536x512 conv weights
    auto cvtTP = [&](int idx, unsigned short* dst) {
        cvtT_bf16_kernel<HH><<<tgP, 256, 0, stream>>>((const float*)d_in[idx], dst);
    };
    auto cvtTC = [&](int idx, unsigned short* dst) {
        cvtT_bf16_kernel<3 * HH><<<tgC, 256, 0, stream>>>((const float*)d_in[idx], dst);
    };
    cvtTP(IN_PP_W1, ppw1);  cvtTP(IN_PP_W2, ppw2);
    cvtTP(IN_EP_W1, epw1);  cvtTP(IN_EP_W2, epw2);
    cvtTP(IN_SP_W1, spw1);  cvtTP(IN_SP_W2, spw2);
    cvtTC(IN_DUR_W1, durw1); cvtTC(IN_DUR_W2, durw2);
    cvtTC(IN_PIT_W1, pitw1); cvtTC(IN_PIT_W2, pitw2);
    cvtTC(IN_ENG_W1, engw1); cvtTC(IN_ENG_W2, engw2);
    // activations
    cvt_bf16_kernel<<<2048, 256, 0, stream>>>(fx, xbf, NACT);
    cvt_bf16_kernel<<<2048, 256, 0, stream>>>((const float*)d_in[IN_EMB], ebf, NACT);

    dim3 ggrid(FF / BN, MROWS / BM);   // (4, 128)
    auto gemmP_bf = [&](const unsigned short* A, const unsigned short* Bw, int biasIdx,
                        unsigned short* oBF) {
        gemm_bf16_kernel<0, HH, 1><<<ggrid, 256, 0, stream>>>(
            A, Bw, (const float*)d_in[biasIdx], nullptr, oBF);
    };
    auto gemmP_f = [&](const unsigned short* A, const unsigned short* Bw, int biasIdx,
                       float* oF) {
        gemm_bf16_kernel<0, HH, 0><<<ggrid, 256, 0, stream>>>(
            A, Bw, (const float*)d_in[biasIdx], oF, nullptr);
    };
    auto gemmC = [&](const unsigned short* A, const unsigned short* Bw, int biasIdx,
                     float* oF) {
        gemm_bf16_kernel<1, 3 * HH, 0><<<ggrid, 256, 0, stream>>>(
            A, Bw, (const float*)d_in[biasIdx], oF, nullptr);
    };
    auto rms = [&](int gIdx, unsigned short* dst) {
        rms_kernel<<<MROWS, 256, 0, stream>>>(accF, (const float*)d_in[gIdx], dst);
    };
    auto dot = [&](const unsigned short* h, int wlIdx, int blIdx, float* o) {
        dot_mask_kernel<<<MROWS / 8, 256, 0, stream>>>(
            h, (const float*)d_in[wlIdx], (const float*)d_in[blIdx], mask, o);
    };

    // ---- duration predictor (input = x) ----
    gemmC(xbf, durw1, IN_DUR_B1, accF);
    rms(IN_DUR_G1, bufB);
    gemmC(bufB, durw2, IN_DUR_B2, accF);
    rms(IN_DUR_G2, bufB);
    dot(bufB, IN_DUR_WL, IN_DUR_BL, outDur);

    // ---- pitch predictor (input = x + proj_pp(embedding)) ----
    gemmP_bf(ebf, ppw1, IN_PP_B1, bufB);
    gemmP_f(bufB, ppw2, IN_PP_B2, accF);
    addx_bf16_kernel<<<2048, 256, 0, stream>>>(fx, accF, bufA, NACT);
    gemmC(bufA, pitw1, IN_PIT_B1, accF);
    rms(IN_PIT_G1, bufB);
    gemmC(bufB, pitw2, IN_PIT_B2, accF);
    rms(IN_PIT_G2, bufB);
    dot(bufB, IN_PIT_WL, IN_PIT_BL, outPit);

    // ---- energy predictor (input = x + proj_ep(embedding)) ----
    gemmP_bf(ebf, epw1, IN_EP_B1, bufB);
    gemmP_f(bufB, epw2, IN_EP_B2, accF);
    addx_bf16_kernel<<<2048, 256, 0, stream>>>(fx, accF, bufA, NACT);
    gemmC(bufA, engw1, IN_ENG_B1, accF);
    rms(IN_ENG_G1, bufB);
    gemmC(bufB, engw2, IN_ENG_B2, accF);
    rms(IN_ENG_G2, bufB);
    dot(bufB, IN_ENG_WL, IN_ENG_BL, outEng);

    // ---- sp projection + xo compose ----
    gemmP_bf(ebf, spw1, IN_SP_B1, bufB);
    gemmP_f(bufB, spw2, IN_SP_B2, accF);
    compose_xo_kernel<<<MROWS, 256, 0, stream>>>(
        fx, accF,
        (const float*)d_in[IN_PITT], (const float*)d_in[IN_ENGT],
        (const float*)d_in[IN_PBINS], (const float*)d_in[IN_EBINS],
        (const float*)d_in[IN_PEMB], (const float*)d_in[IN_EEMB],
        xoF);

    // ---- length regulate ----
    cumsum_kernel<<<BB, LL, 0, stream>>>(durT, cum, outMelLen);
    lenreg_kernel<<<BB * TT, 256, 0, stream>>>(xoF, cum, outMel0);
}
